// MultiHeadAttention_85555748536786
// MI455X (gfx1250) — compile-verified, matrix operand order checked by reading, not run on hardware
//
#include <hip/hip_runtime.h>
#include <hip/hip_bf16.h>

// ---------------- CDNA5 (gfx1250) wave32 WMMA MHA ----------------
// All GEMMs on V_WMMA_F32_16X16X32_BF16 (fp32 accum). K-loops are fully
// unrolled straight-line SSA: the scheduler hoists global_load_b128 ahead
// of the WMMAs (pressure-bounded), and accumulators coalesce in place
// (no v_mov shuffling between loop iterations).

typedef __attribute__((ext_vector_type(16))) __bf16 v16bf;
typedef __attribute__((ext_vector_type(8)))  float  v8f;
typedef __attribute__((ext_vector_type(4)))  float  f32x4;

#define D_MODEL 1024
#define NHEAD   16
#define DK      64
#define BATCH   8
#define SEQ     1024

__device__ inline v8f wmma_bf16(v16bf a, v16bf b, v8f c) {
  return __builtin_amdgcn_wmma_f32_16x16x32_bf16(false, a, false, b, (short)0, c,
                                                 false, false);
}

// ---- fragment loaders -------------------------------------------------------
// ISA 7.12.2, 16-bit A/B 16x32 tile: lane<16 holds K={0..7,16..23} of row
// (lane&15); lane>=16 holds K={8..15,24..31}. Two 16B loads per lane.
__device__ inline v16bf load_frag_bf16(const __bf16* __restrict__ base, int ld) {
  const int lane = threadIdx.x & 31;
  const int r    = lane & 15;
  const int kb   = (lane < 16) ? 0 : 8;
  const __bf16* p = base + (size_t)r * ld + kb;
  union { v16bf v; f32x4 q[2]; } u;
  u.q[0] = *reinterpret_cast<const f32x4*>(p);
  u.q[1] = *reinterpret_cast<const f32x4*>(p + 16);
  return u.v;
}

// fp32 source; convert in-register (v_cvt_pk_bf16_f32).
__device__ inline v16bf load_frag_f32cvt(const float* __restrict__ base, int ld) {
  const int lane = threadIdx.x & 31;
  const int r    = lane & 15;
  const int kb   = (lane < 16) ? 0 : 8;
  const float* p = base + (size_t)r * ld + kb;
  union { float f[16]; f32x4 q[4]; } u;
  u.q[0] = *reinterpret_cast<const f32x4*>(p);
  u.q[1] = *reinterpret_cast<const f32x4*>(p + 4);
  u.q[2] = *reinterpret_cast<const f32x4*>(p + 16);
  u.q[3] = *reinterpret_cast<const f32x4*>(p + 20);
  v16bf v;
#pragma unroll
  for (int i = 0; i < 16; ++i) v[i] = (__bf16)u.f[i];
  return v;
}

// ---- accumulator stores (C/D: VGPR i = row (lane<16 ? i : i+8), col lane&15)
__device__ inline void store_tile_f32(float* __restrict__ base, int ld, v8f c) {
  const int lane = threadIdx.x & 31;
  const int n = lane & 15, mo = (lane & 16) >> 1;
#pragma unroll
  for (int i = 0; i < 8; ++i) base[(size_t)(mo + i) * ld + n] = c[i];
}

__device__ inline void store_tile_bf16(__bf16* __restrict__ base, int ld, v8f c) {
  const int lane = threadIdx.x & 31;
  const int n = lane & 15, mo = (lane & 16) >> 1;
#pragma unroll
  for (int i = 0; i < 8; ++i) base[(size_t)(mo + i) * ld + n] = (__bf16)c[i];
}

// transposed: tile element (m,n) -> base[n*ldt + m]; 8 contiguous bf16 = 16B
__device__ inline void store_tile_bf16_tr(__bf16* __restrict__ base, int ldt, v8f c) {
  const int lane = threadIdx.x & 31;
  const int n = lane & 15, mo = (lane & 16) >> 1;
  union { __bf16 h[8]; f32x4 q; } u;
#pragma unroll
  for (int i = 0; i < 8; ++i) u.h[i] = (__bf16)c[i];
  *reinterpret_cast<f32x4*>(base + (size_t)n * ldt + mo) = u.q;
}

// ---- wave-tile fragment bundle: 32 rows (2 A frags) x 64 cols (4 B frags) --
struct Frag32x64 { v16bf a0, a1, b[4]; };

__device__ inline Frag32x64 load_bundle_bf16(const __bf16* __restrict__ A0,
                                             const __bf16* __restrict__ A1,
                                             int lda,
                                             const __bf16* __restrict__ B,
                                             int ldb, int k0) {
  Frag32x64 f;
  f.a0 = load_frag_bf16(A0 + k0, lda);
  f.a1 = load_frag_bf16(A1 + k0, lda);
#pragma unroll
  for (int t = 0; t < 4; ++t)
    f.b[t] = load_frag_bf16(B + (size_t)(t * 16) * ldb + k0, ldb);
  return f;
}

__device__ inline Frag32x64 load_bundle_f32A(const float* __restrict__ A0,
                                             const float* __restrict__ A1,
                                             int lda,
                                             const __bf16* __restrict__ B,
                                             int ldb, int k0) {
  Frag32x64 f;
  f.a0 = load_frag_f32cvt(A0 + k0, lda);
  f.a1 = load_frag_f32cvt(A1 + k0, lda);
#pragma unroll
  for (int t = 0; t < 4; ++t)
    f.b[t] = load_frag_bf16(B + (size_t)(t * 16) * ldb + k0, ldb);
  return f;
}

__device__ inline void mma_bundle(v8f (&acc)[8], const Frag32x64& f) {
#pragma unroll
  for (int t = 0; t < 4; ++t) {
    acc[t]     = wmma_bf16(f.a0, f.b[t], acc[t]);
    acc[4 + t] = wmma_bf16(f.a1, f.b[t], acc[4 + t]);
  }
}

// ---- kernel 0: one-time fp32 -> bf16 conversion (8 elems / thread) ---------
__global__ __launch_bounds__(256) void cvt_bf16_kernel(
    const float* __restrict__ src, __bf16* __restrict__ dst, int n8) {
  const int i = blockIdx.x * 256 + threadIdx.x;
  if (i >= n8) return;
  const f32x4* s = reinterpret_cast<const f32x4*>(src) + 2 * (size_t)i;
  const f32x4 a = s[0], b = s[1];
  union { __bf16 h[8]; f32x4 q; } u;
#pragma unroll
  for (int j = 0; j < 4; ++j) { u.h[j] = (__bf16)a[j]; u.h[4 + j] = (__bf16)b[j]; }
  *(reinterpret_cast<f32x4*>(dst) + i) = u.q;
}

// ---- kernel 1: X @ W^T + b  (bf16 in/out, per-head layout) -----------------
// grid (32, 16): x = 256-row block, y = head; wave tile = 32 rows x 64 cols.
__global__ __launch_bounds__(256) void proj_kernel(
    const __bf16* __restrict__ X, const __bf16* __restrict__ W,
    const float* __restrict__ bias, __bf16* __restrict__ out,
    int storeTransposed) {
  const int wave = threadIdx.x >> 5;
  const int lane = threadIdx.x & 31;
  const int m0   = blockIdx.x * 256 + wave * 32;  // row in [0, 8192)
  const int h    = blockIdx.y;
  const int b    = m0 >> 10;
  const int s0   = m0 & (SEQ - 1);

  const __bf16* X0 = X + (size_t)m0 * D_MODEL;
  const __bf16* X1 = X0 + (size_t)16 * D_MODEL;
  const __bf16* Wb = W + (size_t)h * DK * D_MODEL;

  v8f acc[8] = {};
#pragma unroll
  for (int k0 = 0; k0 < D_MODEL; k0 += 32) {   // straight-line: loads hoist,
    Frag32x64 f = load_bundle_bf16(X0, X1, D_MODEL, Wb, D_MODEL, k0);
    mma_bundle(acc, f);                        // acc coalesces in place
  }

  const int n = lane & 15;
#pragma unroll
  for (int t = 0; t < 4; ++t) {
    const float bv = bias[h * DK + t * 16 + n];
#pragma unroll
    for (int i = 0; i < 8; ++i) { acc[t][i] += bv; acc[4 + t][i] += bv; }
  }
  const size_t bh = (size_t)b * NHEAD + h;
  if (!storeTransposed) {                       // [B,H,S,64]
    __bf16* dst = out + (bh * SEQ + s0) * DK;
#pragma unroll
    for (int t = 0; t < 4; ++t) {
      store_tile_bf16(dst + t * 16, DK, acc[t]);
      store_tile_bf16(dst + (size_t)16 * DK + t * 16, DK, acc[4 + t]);
    }
  } else {                                      // V -> [B,H,64,S]
    __bf16* dst = out + bh * (size_t)DK * SEQ + s0;
#pragma unroll
    for (int t = 0; t < 4; ++t) {
      store_tile_bf16_tr(dst + (size_t)(t * 16) * SEQ, SEQ, acc[t]);
      store_tile_bf16_tr(dst + (size_t)(t * 16) * SEQ + 16, SEQ, acc[4 + t]);
    }
  }
}

// ---- kernel 2: scores = softmax(Q K^T / 8), writes fp32 attention ----------
// One WG per (b,h,16 q-rows). Wave w owns score columns [w*128, w*128+128).
__global__ __launch_bounds__(256) void scores_softmax_kernel(
    const __bf16* __restrict__ Qh, const __bf16* __restrict__ Kh,
    float* __restrict__ attn) {
  extern __shared__ float lds[];           // [16][SEQ+8] fp32  (~66 KB)
  const int LDP  = SEQ + 8;
  const int qt   = blockIdx.x & 63;
  const int bh   = blockIdx.x >> 6;
  const int q0   = qt * 16;
  const int wave = threadIdx.x >> 5;
  const int lane = threadIdx.x & 31;

  const __bf16* Qbase = Qh + ((size_t)bh * SEQ + q0) * DK;
  const v16bf a0 = load_frag_bf16(Qbase, DK);
  const v16bf a1 = load_frag_bf16(Qbase + 32, DK);

  const __bf16* Kcol = Kh + ((size_t)bh * SEQ + wave * 128) * DK;
#pragma unroll
  for (int t = 0; t < 8; ++t) {              // straight-line: loads hoist
    const __bf16* Kt = Kcol + (size_t)(t * 16) * DK;
    v8f c = {};
    c = wmma_bf16(a0, load_frag_bf16(Kt, DK), c);
    c = wmma_bf16(a1, load_frag_bf16(Kt + 32, DK), c);
    const int ki0 = wave * 128 + t * 16;
    const int n = lane & 15, mo = (lane & 16) >> 1;
#pragma unroll
    for (int i = 0; i < 8; ++i)
      lds[(mo + i) * LDP + ki0 + n] = c[i] * 0.125f;  // 1/sqrt(64)
  }
  __syncthreads();

  // softmax: wave owns rows 2w, 2w+1; lanes stride columns (coalesced 128B).
#pragma unroll
  for (int rr = 0; rr < 2; ++rr) {
    const int r = wave * 2 + rr;
    float* row  = lds + r * LDP;
    float m = -3.4e38f;
    for (int c = lane; c < SEQ; c += 32) m = fmaxf(m, row[c]);
#pragma unroll
    for (int off = 16; off > 0; off >>= 1) m = fmaxf(m, __shfl_xor(m, off, 32));
    float s = 0.f;
    for (int c = lane; c < SEQ; c += 32) {
      const float e = __expf(row[c] - m);
      row[c] = e;
      s += e;
    }
#pragma unroll
    for (int off = 16; off > 0; off >>= 1) s += __shfl_xor(s, off, 32);
    const float inv = 1.0f / s;
    float* out = attn + ((size_t)bh * SEQ + q0 + r) * SEQ;
    for (int c = lane; c < SEQ; c += 32)
      __builtin_nontemporal_store(row[c] * inv, out + c);  // stream past L2
  }
}

// ---- kernel 3: ctx = P @ V   (P fp32 read+cvt, Vt bf16) --------------------
// grid = B*H*(S/256); wave tile = 32 rows x 64 cols (full Dk).
__global__ __launch_bounds__(256) void ctx_kernel(
    const float* __restrict__ attn, const __bf16* __restrict__ Vt,
    __bf16* __restrict__ ctx) {
  const int bh   = blockIdx.x >> 2;
  const int st   = blockIdx.x & 3;
  const int wave = threadIdx.x >> 5;
  const int s0   = st * 256 + wave * 32;
  const int b    = bh >> 4, h = bh & 15;

  const float*  A0 = attn + ((size_t)bh * SEQ + s0) * SEQ;
  const float*  A1 = A0 + (size_t)16 * SEQ;
  const __bf16* Vb = Vt + (size_t)bh * DK * SEQ;

  v8f acc[8] = {};
#pragma unroll
  for (int kk = 0; kk < SEQ; kk += 32) {
    Frag32x64 f = load_bundle_f32A(A0, A1, SEQ, Vb, SEQ, kk);
    mma_bundle(acc, f);
  }

  __bf16* dst = ctx + ((size_t)b * SEQ + s0) * D_MODEL + h * DK;
#pragma unroll
  for (int t = 0; t < 4; ++t) {
    store_tile_bf16(dst + t * 16, D_MODEL, acc[t]);
    store_tile_bf16(dst + (size_t)16 * D_MODEL + t * 16, D_MODEL, acc[4 + t]);
  }
}

// ---- kernel 4: out = ctx @ Wo^T + bo  (fp32 out) ---------------------------
// grid (32, 16); wave tile = 32 rows x 64 cols.
__global__ __launch_bounds__(256) void outproj_kernel(
    const __bf16* __restrict__ ctx, const __bf16* __restrict__ Wo,
    const float* __restrict__ bo, float* __restrict__ out) {
  const int wave = threadIdx.x >> 5;
  const int lane = threadIdx.x & 31;
  const int m0   = blockIdx.x * 256 + wave * 32;
  const int n0   = blockIdx.y * 64;

  const __bf16* A0 = ctx + (size_t)m0 * D_MODEL;
  const __bf16* A1 = A0 + (size_t)16 * D_MODEL;
  const __bf16* Wb = Wo + (size_t)n0 * D_MODEL;

  v8f acc[8] = {};
#pragma unroll
  for (int k0 = 0; k0 < D_MODEL; k0 += 32) {
    Frag32x64 f = load_bundle_bf16(A0, A1, D_MODEL, Wb, D_MODEL, k0);
    mma_bundle(acc, f);
  }

  const int n = lane & 15;
#pragma unroll
  for (int t = 0; t < 4; ++t) {
    const float bv = bo[n0 + t * 16 + n];
#pragma unroll
    for (int i = 0; i < 8; ++i) { acc[t][i] += bv; acc[4 + t][i] += bv; }
  }
  float* dst = out + (size_t)m0 * D_MODEL + n0;
#pragma unroll
  for (int t = 0; t < 4; ++t) {
    store_tile_f32(dst + t * 16, D_MODEL, acc[t]);
    store_tile_f32(dst + (size_t)16 * D_MODEL + t * 16, D_MODEL, acc[4 + t]);
  }
}

// ---------------------------------------------------------------------------
extern "C" void kernel_launch(void* const* d_in, const int* in_sizes, int n_in,
                              void* d_out, int out_size, void* d_ws, size_t ws_size,
                              hipStream_t stream) {
  (void)in_sizes; (void)n_in; (void)out_size; (void)ws_size;
  const float* q  = (const float*)d_in[0];
  const float* k  = (const float*)d_in[1];
  const float* v  = (const float*)d_in[2];
  const float* Wq = (const float*)d_in[3];
  const float* bq = (const float*)d_in[4];
  const float* Wk = (const float*)d_in[5];
  const float* bk = (const float*)d_in[6];
  const float* Wv = (const float*)d_in[7];
  const float* bv = (const float*)d_in[8];
  const float* Wo = (const float*)d_in[9];
  const float* bo = (const float*)d_in[10];

  float* out  = (float*)d_out;                               // [B,S,D] fp32
  float* attn = out + (size_t)BATCH * SEQ * D_MODEL;         // [B,H,S,S] fp32

  const size_t actE = (size_t)BATCH * SEQ * D_MODEL;         // 8 Mi elems
  const size_t wE   = (size_t)D_MODEL * D_MODEL;             // 1 Mi elems
  __bf16* qb  = (__bf16*)d_ws;
  __bf16* kb  = qb  + actE;
  __bf16* vb  = kb  + actE;
  __bf16* Wqb = vb  + actE;
  __bf16* Wkb = Wqb + wE;
  __bf16* Wvb = Wkb + wE;
  __bf16* Wob = Wvb + wE;
  __bf16* Qh  = Wob + wE;                 // [B,H,S,64]
  __bf16* Kh  = Qh  + actE;               // [B,H,S,64]
  __bf16* Vt  = Kh  + actE;               // [B,H,64,S]
  __bf16* ctx = Vt  + actE;               // [B,S,1024]   (~126 MB total ws)

  const dim3 block(256);

  const int actG = (int)(actE / 8 / 256), wG = (int)(wE / 8 / 256);
  cvt_bf16_kernel<<<actG, block, 0, stream>>>(q,  qb,  (int)(actE / 8));
  cvt_bf16_kernel<<<actG, block, 0, stream>>>(k,  kb,  (int)(actE / 8));
  cvt_bf16_kernel<<<actG, block, 0, stream>>>(v,  vb,  (int)(actE / 8));
  cvt_bf16_kernel<<<wG,   block, 0, stream>>>(Wq, Wqb, (int)(wE / 8));
  cvt_bf16_kernel<<<wG,   block, 0, stream>>>(Wk, Wkb, (int)(wE / 8));
  cvt_bf16_kernel<<<wG,   block, 0, stream>>>(Wv, Wvb, (int)(wE / 8));
  cvt_bf16_kernel<<<wG,   block, 0, stream>>>(Wo, Wob, (int)(wE / 8));

  const dim3 gridP(32, 16);
  proj_kernel<<<gridP, block, 0, stream>>>(qb, Wqb, bq, Qh, 0);
  proj_kernel<<<gridP, block, 0, stream>>>(kb, Wkb, bk, Kh, 0);
  proj_kernel<<<gridP, block, 0, stream>>>(vb, Wvb, bv, Vt, 1);

  const int ldsBytes = (SEQ + 8) * 16 * (int)sizeof(float);  // 66048 B
  scores_softmax_kernel<<<BATCH * NHEAD * (SEQ / 16), block, ldsBytes, stream>>>(
      Qh, Kh, attn);

  ctx_kernel<<<BATCH * NHEAD * (SEQ / 256), block, 0, stream>>>(attn, Vt, ctx);

  outproj_kernel<<<dim3(32, 16), block, 0, stream>>>(ctx, Wob, bo, out);
}
